// Net_77137612636876
// MI455X (gfx1250) — compile-verified, matrix operand order checked by reading, not run on hardware
//
#include <hip/hip_runtime.h>
#include <hip/hip_bf16.h>

typedef __attribute__((ext_vector_type(16))) _Float16 v16h;
typedef __attribute__((ext_vector_type(8)))  _Float16 v8h;
typedef __attribute__((ext_vector_type(8)))  float    v8f;

#define BATCH  2048
#define TSTEPS 96
#define DIN    64
#define HID    256
#define NK     20

__device__ __forceinline__ float fsig(float x)  { return 1.f / (1.f + __expf(-x)); }
__device__ __forceinline__ float ftanh(float x) { float e = __expf(2.f * x); return (e - 1.f) / (e + 1.f); }
__device__ __forceinline__ float fsoftplus(float x) { return (x > 20.f) ? x : logf(1.f + __expf(x)); }

__device__ __forceinline__ v16h ld16(const _Float16* p) { return *(const v16h*)p; }

__device__ __forceinline__ v8f wmma_f16(v16h a, v16h b, v8f c) {
  // D = A(16x32 f16) * B(32x16 f16) + C(16x16 f32)
  return __builtin_amdgcn_wmma_f32_16x16x32_f16(false, a, false, b, (short)0, c, false, false);
}

// B-fragment from a row-major [outCols][ldk] f16 weight matrix.
// lane ln (0..15) supplies column N = coltile*16 + ln, hf selects K half.
__device__ __forceinline__ v16h ldB(const _Float16* __restrict__ W, int ldk,
                                    int coltile, int kt, int hf, int ln) {
  return ld16(W + (size_t)(coltile * 16 + ln) * ldk + kt * 32 + hf * 16);
}

// ---------------- prep kernels ----------------

__global__ void k_zero(float* o) { o[0] = 0.f; }

__global__ void k_cvt(const float* __restrict__ s, _Float16* __restrict__ d, int n) {
  int i = blockIdx.x * blockDim.x + threadIdx.x;
  if (i < n) d[i] = (_Float16)s[i];
}

__global__ void k_build_wp(const float* __restrict__ wpb, const float* __restrict__ wpg,
                           _Float16* __restrict__ wph) {
  int i = blockIdx.x * blockDim.x + threadIdx.x;  // 32*512
  if (i >= 32 * 512) return;
  int row = i >> 9, col = i & 511;
  float v = 0.f;
  if (row == 0)        v = wpb[col];
  else if (row <= NK)  v = wpg[(row - 1) * 512 + col];
  wph[i] = (_Float16)v;
}

__global__ void k_build_bias(const float* bi0, const float* bh0,
                             const float* bi1, const float* bh1,
                             const float* bpb, const float* bpg,
                             float* bias0, float* bias1, float* biasP) {
  int i = blockIdx.x * blockDim.x + threadIdx.x;
  if (i < 1024) bias0[i] = bi0[i] + bh0[i];
  else if (i < 2048) { int j = i - 1024; bias1[j] = bi1[j] + bh1[j]; }
  else if (i < 2080) {
    int j = i - 2048;
    float v = 0.f;
    if (j == 0)       v = bpb[0];
    else if (j <= NK) v = bpg[j - 1];
    biasP[j] = v;
  }
}

// ---------------- main persistent LSTM+CRPS kernel ----------------
// One wave32 per block owns a 16-row batch slab for all 96 timesteps.

__global__ void __launch_bounds__(32)
k_lstm_crps(const _Float16* __restrict__ Xh,
            const _Float16* __restrict__ Wi0h, const _Float16* __restrict__ Wh0h,
            const _Float16* __restrict__ Wi1h, const _Float16* __restrict__ Wh1h,
            const _Float16* __restrict__ Wph,
            const float* __restrict__ bias0, const float* __restrict__ bias1,
            const float* __restrict__ biasP,
            const float* __restrict__ ylab, float* __restrict__ out) {
  __shared__ __align__(32) _Float16 h0s[2][16][HID];   // 16 KB, double buffered
  __shared__ __align__(32) _Float16 h1s[2][16][HID];   // 16 KB
  __shared__ __align__(16) float    c0s[16][32][8];    // 16 KB, lane-private c state
  __shared__ __align__(16) float    c1s[16][32][8];    // 16 KB
  __shared__ __align__(16) float    projs[16][32];     // 2 KB

  const int lane    = threadIdx.x;
  const int rowbase = blockIdx.x * 16;
  const int hf = lane >> 4;    // which K-half / row-half this lane covers
  const int ln = lane & 15;

  // zero recurrent state
  for (int i = lane; i < 2 * 16 * HID; i += 32) {
    ((_Float16*)h0s)[i] = (_Float16)0.f;
    ((_Float16*)h1s)[i] = (_Float16)0.f;
  }
  for (int i = lane; i < 16 * 32 * 8; i += 32) {
    ((float*)c0s)[i] = 0.f;
    ((float*)c1s)[i] = 0.f;
  }
  __syncthreads();

  float lossAcc = 0.f;
  int p = 0;

#pragma unroll 1
  for (int t = 0; t < TSTEPS; ++t) {
    const int q = p ^ 1;

    // x A-fragments for this step: lane ln covers row rowbase+ln
    const _Float16* xbase = Xh + ((size_t)(rowbase + ln) * TSTEPS + t) * DIN + hf * 16;
    v16h ax0 = ld16(xbase);
    v16h ax1 = ld16(xbase + 32);

    // h0_{t-1} A-fragments
    v16h a0[8];
#pragma unroll
    for (int kt = 0; kt < 8; ++kt)
      a0[kt] = ld16(&h0s[p][ln][kt * 32 + hf * 16]);

    // ---------------- layer 0 ----------------
#pragma unroll 1
    for (int j = 0; j < 16; ++j) {
      v8f acc[4];
#pragma unroll
      for (int g = 0; g < 4; ++g) {
        float bv = bias0[g * 256 + j * 16 + ln];
#pragma unroll
        for (int v = 0; v < 8; ++v) acc[g][v] = bv;
      }
#pragma unroll
      for (int kt = 0; kt < 2; ++kt) {
        v16h a = kt ? ax1 : ax0;
#pragma unroll
        for (int g = 0; g < 4; ++g)
          acc[g] = wmma_f16(a, ldB(Wi0h, DIN, g * 16 + j, kt, hf, ln), acc[g]);
      }
#pragma unroll
      for (int kt = 0; kt < 8; ++kt) {
#pragma unroll
        for (int g = 0; g < 4; ++g)
          acc[g] = wmma_f16(a0[kt], ldB(Wh0h, HID, g * 16 + j, kt, hf, ln), acc[g]);
      }
      float* cp = &c0s[j][lane][0];
#pragma unroll
      for (int v = 0; v < 8; ++v) {
        float iv = fsig(acc[0][v]);
        float fv = fsig(acc[1][v]);
        float gv = ftanh(acc[2][v]);
        float ov = fsig(acc[3][v]);
        float c2 = fv * cp[v] + iv * gv;
        cp[v] = c2;
        h0s[q][hf * 8 + v][j * 16 + ln] = (_Float16)(ov * ftanh(c2));
      }
    }
    __syncthreads();   // h0_t visible cross-lane (1-wave WG: ~S_NOP, compiler fence)

    // ---------------- layer 1 ----------------
    v16h a1a[8], a1b[8];
#pragma unroll
    for (int kt = 0; kt < 8; ++kt) {
      a1a[kt] = ld16(&h0s[q][ln][kt * 32 + hf * 16]);
      a1b[kt] = ld16(&h1s[p][ln][kt * 32 + hf * 16]);
    }
#pragma unroll 1
    for (int j = 0; j < 16; ++j) {
      v8f acc[4];
#pragma unroll
      for (int g = 0; g < 4; ++g) {
        float bv = bias1[g * 256 + j * 16 + ln];
#pragma unroll
        for (int v = 0; v < 8; ++v) acc[g][v] = bv;
      }
#pragma unroll
      for (int kt = 0; kt < 8; ++kt) {
#pragma unroll
        for (int g = 0; g < 4; ++g)
          acc[g] = wmma_f16(a1a[kt], ldB(Wi1h, HID, g * 16 + j, kt, hf, ln), acc[g]);
      }
#pragma unroll
      for (int kt = 0; kt < 8; ++kt) {
#pragma unroll
        for (int g = 0; g < 4; ++g)
          acc[g] = wmma_f16(a1b[kt], ldB(Wh1h, HID, g * 16 + j, kt, hf, ln), acc[g]);
      }
      float* cp = &c1s[j][lane][0];
#pragma unroll
      for (int v = 0; v < 8; ++v) {
        float iv = fsig(acc[0][v]);
        float fv = fsig(acc[1][v]);
        float gv = ftanh(acc[2][v]);
        float ov = fsig(acc[3][v]);
        float c2 = fv * cp[v] + iv * gv;
        cp[v] = c2;
        h1s[q][hf * 8 + v][j * 16 + ln] = (_Float16)(ov * ftanh(c2));
      }
    }
    __syncthreads();

    // ---------------- projection via WMMA: hp[16x512] @ Wp^T[512x32] ----------------
    v8f pacc[2];
#pragma unroll
    for (int tt = 0; tt < 2; ++tt) {
      float bv = biasP[tt * 16 + ln];
#pragma unroll
      for (int v = 0; v < 8; ++v) pacc[tt][v] = bv;
    }
#pragma unroll 1
    for (int kt = 0; kt < 16; ++kt) {
      int hb = kt * 16 + hf * 8;             // h-column base; hp interleaves h0/h1
      v8h e0 = *(const v8h*)&h0s[q][ln][hb];
      v8h e1 = *(const v8h*)&h1s[q][ln][hb];
      v16h a;
#pragma unroll
      for (int i = 0; i < 8; ++i) { a[2 * i] = e0[i]; a[2 * i + 1] = e1[i]; }
      pacc[0] = wmma_f16(a, ld16(Wph + (size_t)ln * 512 + kt * 32 + hf * 16), pacc[0]);
      pacc[1] = wmma_f16(a, ld16(Wph + (size_t)(16 + ln) * 512 + kt * 32 + hf * 16), pacc[1]);
    }
#pragma unroll
    for (int v = 0; v < 8; ++v) {
      projs[hf * 8 + v][ln]      = pacc[0][v];
      projs[hf * 8 + v][16 + ln] = pacc[1][v];
    }
    __syncthreads();

    // ---------------- CRPS spline loss (one row per lane<16) ----------------
    if (lane < 16) {
      float y = ylab[(size_t)(rowbase + ln) * TSTEPS + t];
      float beta0 = fsoftplus(projs[ln][0]);
      float gam[NK];
#pragma unroll
      for (int k = 0; k < NK; ++k) gam[k] = fsoftplus(projs[ln][1 + k]);

      float bb[NK];
      float bt_prev = 0.f, prevg = beta0, ssum = 0.f;
#pragma unroll
      for (int k = 0; k < NK; ++k) {
        float btk = (gam[k] - prevg) * (0.5f * NK);
        prevg = gam[k];
        bb[k] = btk - bt_prev;
        bt_prev = btk;
        if (k < NK - 1) ssum += bb[k];
      }
      bb[NK - 1] = gam[NK - 1] - ssum;

      const float invK = 1.f / NK;
      float A = 0.f, Bq = beta0, Cq = -y;
      float Fprev = 0.f, bestAbs = 3.4e38f, alphaKnot = 0.f;
      unsigned almask = 0;
#pragma unroll
      for (int i = 0; i < NK; ++i) {
        float ksi_i = i * invK;
        float diff = y - Fprev;
        float ad = fabsf(diff);
        if (ad < bestAbs) { bestAbs = ad; alphaKnot = ksi_i; }
        if (diff > 0.f) {
          almask |= (1u << i);
          A  += bb[i];
          Bq -= 2.f * bb[i] * ksi_i;
          Cq += bb[i] * ksi_i * ksi_i;
        }
        float kc = (i + 1) * invK;        // CDF at ksi_cum[i] -> next knot
        float f = kc * beta0;
#pragma unroll
        for (int k = 0; k < NK; ++k) {
          float d = fmaxf(kc - k * invK, 0.f);
          f += d * d * bb[k];
        }
        Fprev = f;
      }
      float disc = Bq * Bq - 4.f * A * Cq;
      float alpha;
      if (A != 0.f && disc >= 0.f)  alpha = (-Bq + sqrtf(disc)) / (2.f * A);
      else if (A == 0.f)            alpha = -Cq / (Bq == 0.f ? 1.f : Bq);
      else                          alpha = alphaKnot;

      float crps = -y * (1.f - 2.f * alpha) + beta0 * (1.f / 3.f - alpha * alpha);
#pragma unroll
      for (int k = 0; k < NK; ++k) {
        float ksi_k = k * invK;
        float om = 1.f - ksi_k;
        crps += bb[k] * (1.f / 6.f) * om * om * om * om;
        if (almask & (1u << k)) {
          float dk = alpha - ksi_k;
          crps -= (2.f / 3.f) * bb[k] * dk * dk * dk;
        }
      }
      lossAcc += crps;
    }
    __syncthreads();
    p = q;
  }

  // reduce 16 active lanes, one atomic per block
#pragma unroll
  for (int off = 16; off > 0; off >>= 1)
    lossAcc += __shfl_down(lossAcc, off, 32);
  if (lane == 0) atomicAdd(out, lossAcc * (1.f / BATCH));
}

// ---------------- host-side launch ----------------

extern "C" void kernel_launch(void* const* d_in, const int* in_sizes, int n_in,
                              void* d_out, int out_size, void* d_ws, size_t ws_size,
                              hipStream_t stream) {
  const float* X   = (const float*)d_in[0];
  const float* Y   = (const float*)d_in[1];
  const float* Wi0 = (const float*)d_in[2];
  const float* Wh0 = (const float*)d_in[3];
  const float* Wi1 = (const float*)d_in[4];
  const float* Wh1 = (const float*)d_in[5];
  const float* bi0 = (const float*)d_in[6];
  const float* bh0 = (const float*)d_in[7];
  const float* bi1 = (const float*)d_in[8];
  const float* bh1 = (const float*)d_in[9];
  const float* Wpb = (const float*)d_in[10];
  const float* bpb = (const float*)d_in[11];
  const float* Wpg = (const float*)d_in[12];
  const float* bpg = (const float*)d_in[13];
  float* out = (float*)d_out;

  char* ws = (char*)d_ws;
  size_t off = 0;
  auto alloc = [&](size_t bytes) -> char* {
    char* p = ws + off;
    off += (bytes + 255) & ~(size_t)255;
    return p;
  };
  _Float16* Wh0h = (_Float16*)alloc((size_t)1024 * 256 * 2);
  _Float16* Wi1h = (_Float16*)alloc((size_t)1024 * 256 * 2);
  _Float16* Wh1h = (_Float16*)alloc((size_t)1024 * 256 * 2);
  _Float16* Wi0h = (_Float16*)alloc((size_t)1024 * 64 * 2);
  _Float16* Wph  = (_Float16*)alloc((size_t)32 * 512 * 2);
  float*    bias0 = (float*)alloc(1024 * 4);
  float*    bias1 = (float*)alloc(1024 * 4);
  float*    biasP = (float*)alloc(32 * 4);
  _Float16* Xh    = (_Float16*)alloc((size_t)BATCH * TSTEPS * DIN * 2);

  k_zero<<<1, 1, 0, stream>>>(out);
  k_cvt<<<(1024 * 256 + 255) / 256, 256, 0, stream>>>(Wh0, Wh0h, 1024 * 256);
  k_cvt<<<(1024 * 256 + 255) / 256, 256, 0, stream>>>(Wi1, Wi1h, 1024 * 256);
  k_cvt<<<(1024 * 256 + 255) / 256, 256, 0, stream>>>(Wh1, Wh1h, 1024 * 256);
  k_cvt<<<(1024 * 64 + 255) / 256, 256, 0, stream>>>(Wi0, Wi0h, 1024 * 64);
  int nx = BATCH * TSTEPS * DIN;
  k_cvt<<<(nx + 255) / 256, 256, 0, stream>>>(X, Xh, nx);
  k_build_wp<<<(32 * 512 + 255) / 256, 256, 0, stream>>>(Wpb, Wpg, Wph);
  k_build_bias<<<(2080 + 255) / 256, 256, 0, stream>>>(bi0, bh0, bi1, bh1, bpb, bpg,
                                                       bias0, bias1, biasP);
  k_lstm_crps<<<BATCH / 16, 32, 0, stream>>>(Xh, Wi0h, Wh0h, Wi1h, Wh1h, Wph,
                                             bias0, bias1, biasP, Y, out);
}
